// LinearAttention_74457553044015
// MI455X (gfx1250) — matrix operand order repaired and checked by reading, not compiled, not run on hardware
//
#include <hip/hip_runtime.h>
#include <hip/hip_bf16.h>

// ---------------------------------------------------------------------------
// LinearAttention fused pipeline for MI455X (gfx1250, wave32, WMMA).
//
//  B=16, C=256, HEADS=8, dh=32, n=H*W=4096, qkv rows = 768.
//
//  ws layout (bytes):                              size
//   qkv (f16)   [b][o<768][p<4096]                 16*768*4096*2 = 100663296
//   kmax (f32)  [b*256 + h*32 + d]                 4096*4
//   ksum (f32)  [b*256 + h*32 + d]                 4096*4
//   ctx  (f32)  [(b*8+h)*1024 + d*32 + e]          16*8*1024*4   = 524288
//   W2   (f32)  [(b*256+o)*256 + h*32+d]           16*65536*4    = 4194304
//  total ~ 100.6 MB
// ---------------------------------------------------------------------------

typedef __attribute__((ext_vector_type(16))) _Float16 v16h;
typedef __attribute__((ext_vector_type(8)))  _Float16 v8h;
typedef __attribute__((ext_vector_type(8)))  float    v8f;

#define NPIX 4096

// ---- WMMA fragment helpers (CDNA5 16-bit layouts, CDNA5 ISA) ----
// A (16x32, MxK), row-major tile in LDS with leading dim `ld`:
//   lanes 0-15: row M=lane, K 0..7 (v0..3) and 16..23 (v4..7)
//   lanes 16-31: row M=lane-16, K 8..15 and 24..31
static __device__ inline v16h frag_a(const _Float16* t, int ld, int lane) {
  const int row = lane & 15;
  const int kb  = (lane >> 4) * 8;
  v8h lo = *(const v8h*)(t + row * ld + kb);       // K kb..kb+7
  v8h hi = *(const v8h*)(t + row * ld + kb + 16);  // K 16+kb..16+kb+7
  return __builtin_shufflevector(lo, hi, 0,1,2,3,4,5,6,7,8,9,10,11,12,13,14,15);
}
// B (32x16, KxN), tile stored column-contiguous: Bs[n*ld + k]:
//   lanes 0-15: col N=lane, K 0..7 and 16..23; lanes 16-31: col N=lane-16, K 8..15 and 24..31
static __device__ inline v16h frag_b(const _Float16* t, int ld, int lane) {
  const int col = lane & 15;
  const int kb  = (lane >> 4) * 8;
  v8h lo = *(const v8h*)(t + col * ld + kb);
  v8h hi = *(const v8h*)(t + col * ld + kb + 16);
  return __builtin_shufflevector(lo, hi, 0,1,2,3,4,5,6,7,8,9,10,11,12,13,14,15);
}
static __device__ inline v8f wmma_f16(v16h a, v16h b, v8f c) {
  return __builtin_amdgcn_wmma_f32_16x16x32_f16(false, a, false, b, (short)0, c,
                                                false, false);
}
static __device__ inline v8f vzero8() {
  v8f z = {0.f, 0.f, 0.f, 0.f, 0.f, 0.f, 0.f, 0.f};
  return z;
}

// ---------------------------------------------------------------------------
// K1: qkv[b,o,p] = sum_c w_qkv[o,c] * x[b,c,p]   (store f16)
// grid (32 ntiles, 6 mtiles, 16 b), block 256 (8 waves).
// Workgroup tile 128x128, wave tile 32x64 (2x4 WMMA 16x16 tiles), K-step 32.
// ---------------------------------------------------------------------------
__global__ __launch_bounds__(256)
void qkv_gemm(const float* __restrict__ x, const float* __restrict__ wqkv,
              _Float16* __restrict__ qkv) {
  __shared__ __align__(16) _Float16 As[128 * 32];  // row-major [m][k]
  __shared__ __align__(16) _Float16 Bs[128 * 32];  // col-contig [p][k]
  const int b    = blockIdx.z;
  const int M0   = blockIdx.y * 128;
  const int N0   = blockIdx.x * 128;
  const int tid  = threadIdx.x;
  const int lane = tid & 31;
  const int wave = tid >> 5;
  const int wm   = wave >> 1;  // 0..3
  const int wn   = wave & 1;   // 0..1

  v8f acc[2][4];
#pragma unroll
  for (int m = 0; m < 2; ++m)
#pragma unroll
    for (int n = 0; n < 4; ++n) acc[m][n] = vzero8();

  for (int k0 = 0; k0 < 256; k0 += 32) {
    {  // A tile: 128x32 of w_qkv, fp32 -> f16
      const int row = tid >> 1;
      const int cb  = (tid & 1) * 16;
      const float* src = wqkv + (size_t)(M0 + row) * 256 + k0 + cb;
#pragma unroll
      for (int j = 0; j < 16; ++j) As[row * 32 + cb + j] = (_Float16)src[j];
    }
    {  // B tile: 32 (c) x 128 (p) of x, transposed into [p][k]
      const int kk = tid >> 3;
      const int pb = (tid & 7) * 16;
      const float* src =
          x + (((size_t)b * 256 + k0 + kk) << 12) + N0 + pb;
#pragma unroll
      for (int j = 0; j < 16; ++j) Bs[(pb + j) * 32 + kk] = (_Float16)src[j];
    }
    __syncthreads();

    v16h af[2], bf[4];
    af[0] = frag_a(As + (wm * 32 + 0) * 32, 32, lane);
    af[1] = frag_a(As + (wm * 32 + 16) * 32, 32, lane);
#pragma unroll
    for (int nt = 0; nt < 4; ++nt)
      bf[nt] = frag_b(Bs + (wn * 64 + nt * 16) * 32, 32, lane);
#pragma unroll
    for (int mt = 0; mt < 2; ++mt)
#pragma unroll
      for (int nt = 0; nt < 4; ++nt)
        acc[mt][nt] = wmma_f16(af[mt], bf[nt], acc[mt][nt]);
    __syncthreads();
  }

  // epilogue: D layout: VGPR r -> M = r + 8*(lane>=16), N = lane&15
  const int mrow = (lane >> 4) * 8;
  const int ncol = lane & 15;
#pragma unroll
  for (int mt = 0; mt < 2; ++mt)
#pragma unroll
    for (int nt = 0; nt < 4; ++nt) {
      const int obase = M0 + wm * 32 + mt * 16 + mrow;
      const int p     = N0 + wn * 64 + nt * 16 + ncol;
      _Float16* dst = qkv + (((size_t)b * 768 + obase) << 12) + p;
#pragma unroll
      for (int r = 0; r < 8; ++r)
        dst[(size_t)r << 12] = (_Float16)acc[mt][nt][r];
    }
}

// ---------------------------------------------------------------------------
// K2: per k-row (b,h,d): max over n and sum(exp(k - max)).  grid 4096.
// ---------------------------------------------------------------------------
__global__ __launch_bounds__(256)
void kstat(const _Float16* __restrict__ qkv, float* __restrict__ kmax,
           float* __restrict__ ksum) {
  const int r   = blockIdx.x;  // b = r>>8, (h*32+d) = r&255
  const int b   = r >> 8;
  const int tid = threadIdx.x;
  const _Float16* src = qkv + (((size_t)b * 768 + 256 + (r & 255)) << 12);
  __shared__ float red[256];

  float m = -1e30f;
  for (int i = tid; i < NPIX; i += 256) m = fmaxf(m, (float)src[i]);
  red[tid] = m;
  __syncthreads();
  for (int s = 128; s > 0; s >>= 1) {
    if (tid < s) red[tid] = fmaxf(red[tid], red[tid + s]);
    __syncthreads();
  }
  const float mx = red[0];
  __syncthreads();

  float sum = 0.f;
  for (int i = tid; i < NPIX; i += 256) sum += __expf((float)src[i] - mx);
  red[tid] = sum;
  __syncthreads();
  for (int s = 128; s > 0; s >>= 1) {
    if (tid < s) red[tid] += red[tid + s];
    __syncthreads();
  }
  if (tid == 0) {
    kmax[r] = mx;
    ksum[r] = red[0];
  }
}

// ---------------------------------------------------------------------------
// K3: ctx[b,h,d,e] = sum_n softmax_k[d,n] * v[e,n]/n.  grid 128 (b*8+h).
// WMMA 32x32 output; exp/scale fused into LDS staging; 8 waves each take a
// 32-wide n slice of each 256-wide chunk; deterministic tree reduction.
// ---------------------------------------------------------------------------
__global__ __launch_bounds__(256)
void context_gemm(const _Float16* __restrict__ qkv,
                  const float* __restrict__ kmax,
                  const float* __restrict__ ksum, float* __restrict__ ctx) {
  __shared__ __align__(16) unsigned char smem[32768];
  _Float16* As = (_Float16*)smem;            // 32 x 256 (k_sm), 16KB
  _Float16* Bs = (_Float16*)(smem + 16384);  // 32 x 256 (v/n),  16KB
  float* red = (float*)smem;                 // 8x1024 overlay after loop

  const int bh   = blockIdx.x;
  const int b    = bh >> 3;
  const int h    = bh & 7;
  const int tid  = threadIdx.x;
  const int lane = tid & 31;
  const int wave = tid >> 5;

  const size_t kbase = ((size_t)(b * 768 + 256 + h * 32)) << 12;
  const size_t vbase = ((size_t)(b * 768 + 512 + h * 32)) << 12;
  const int srow = tid >> 3;        // 0..31 (d for k-load, e for v-load)
  const int pb   = (tid & 7) * 32;  // 32 pixels per thread
  const float km = kmax[b * 256 + h * 32 + srow];
  const float kr = 1.0f / ksum[b * 256 + h * 32 + srow];
  const float vs = 1.0f / (float)NPIX;

  v8f acc[2][2];
#pragma unroll
  for (int m = 0; m < 2; ++m)
#pragma unroll
    for (int n = 0; n < 2; ++n) acc[m][n] = vzero8();

  for (int n0 = 0; n0 < NPIX; n0 += 256) {
    const _Float16* ks = qkv + kbase + ((size_t)srow << 12) + n0 + pb;
    const _Float16* vsrc = qkv + vbase + ((size_t)srow << 12) + n0 + pb;
#pragma unroll
    for (int j = 0; j < 32; ++j) {
      As[srow * 256 + pb + j] = (_Float16)(__expf((float)ks[j] - km) * kr);
      Bs[srow * 256 + pb + j] = (_Float16)((float)vsrc[j] * vs);
    }
    __syncthreads();

    const int no = wave * 32;  // this wave's K slice
    v16h a0 = frag_a(As + 0 * 256 + no, 256, lane);
    v16h a1 = frag_a(As + 16 * 256 + no, 256, lane);
    v16h b0 = frag_b(Bs + 0 * 256 + no, 256, lane);
    v16h b1 = frag_b(Bs + 16 * 256 + no, 256, lane);
    acc[0][0] = wmma_f16(a0, b0, acc[0][0]);
    acc[0][1] = wmma_f16(a0, b1, acc[0][1]);
    acc[1][0] = wmma_f16(a1, b0, acc[1][0]);
    acc[1][1] = wmma_f16(a1, b1, acc[1][1]);
    __syncthreads();
  }

  // per-wave partials -> LDS (overlays As/Bs; safe after final barrier)
  const int mrow = (lane >> 4) * 8;
  const int ncol = lane & 15;
#pragma unroll
  for (int mt = 0; mt < 2; ++mt)
#pragma unroll
    for (int nt = 0; nt < 2; ++nt)
#pragma unroll
      for (int r = 0; r < 8; ++r) {
        const int d = mt * 16 + mrow + r;
        const int e = nt * 16 + ncol;
        red[wave * 1024 + d * 32 + e] = acc[mt][nt][r];
      }
  __syncthreads();
#pragma unroll
  for (int j = 0; j < 4; ++j) {
    const int idx = tid * 4 + j;
    float s = 0.f;
#pragma unroll
    for (int w = 0; w < 8; ++w) s += red[w * 1024 + idx];
    ctx[((size_t)bh << 10) + idx] = s;
  }
}

// ---------------------------------------------------------------------------
// K3b: fold context into output weight:
//   W2[b][o][h*32+d] = sum_e w_out[o][h*32+e] * ctx[b,h,d,e].  grid 128.
// ---------------------------------------------------------------------------
__global__ __launch_bounds__(256)
void fold_w2(const float* __restrict__ wout, const float* __restrict__ ctx,
             float* __restrict__ w2) {
  __shared__ float c[1024];
  const int bh  = blockIdx.x;
  const int b   = bh >> 3;
  const int h   = bh & 7;
  const int tid = threadIdx.x;  // == o
#pragma unroll
  for (int j = 0; j < 4; ++j)
    c[tid * 4 + j] = ctx[((size_t)bh << 10) + tid * 4 + j];
  __syncthreads();

  float wrow[32];
  const float* wsrc = wout + (size_t)tid * 256 + h * 32;
#pragma unroll
  for (int e = 0; e < 32; ++e) wrow[e] = wsrc[e];
  float* dst = w2 + ((size_t)(b * 256 + tid)) * 256 + h * 32;
#pragma unroll 4
  for (int d = 0; d < 32; ++d) {
    float s = 0.f;
#pragma unroll
    for (int e = 0; e < 32; ++e) s += wrow[e] * c[d * 32 + e];
    dst[d] = s;
  }
}

// ---------------------------------------------------------------------------
// K4: y[b,o,p] = sum_{hd} W2[b,o,hd] * q_sm[b,hd,p] + b_out[o]
//   q_sm = softmax over d (32 rows per head, fused into B-tile staging) * dh^-.5
// grid (32 ntiles, 2 mtiles, 16 b), block 256. Writes fp32 straight to d_out.
// ---------------------------------------------------------------------------
__global__ __launch_bounds__(256)
void out_gemm(const _Float16* __restrict__ qkv, const float* __restrict__ w2,
              const float* __restrict__ bout, float* __restrict__ y) {
  __shared__ __align__(16) _Float16 As[128 * 32];
  __shared__ __align__(16) _Float16 Bs[128 * 32];  // [p][k] softmaxed q
  __shared__ float Qt[32 * 128];                   // raw q staging
  const int b    = blockIdx.z;
  const int M0   = blockIdx.y * 128;
  const int N0   = blockIdx.x * 128;
  const int tid  = threadIdx.x;
  const int lane = tid & 31;
  const int wave = tid >> 5;
  const int wm   = wave >> 1;
  const int wn   = wave & 1;
  const float qscale = 0.17677669529663687f;  // 32^-0.5

  v8f acc[2][4];
#pragma unroll
  for (int m = 0; m < 2; ++m)
#pragma unroll
    for (int n = 0; n < 4; ++n) acc[m][n] = vzero8();

  for (int k0 = 0; k0 < 256; k0 += 32) {  // k0/32 == head index
    {  // A tile from W2 (fp32 -> f16)
      const int row = tid >> 1;
      const int cb  = (tid & 1) * 16;
      const float* src = w2 + ((size_t)(b * 256 + M0 + row)) * 256 + k0 + cb;
#pragma unroll
      for (int j = 0; j < 16; ++j) As[row * 32 + cb + j] = (_Float16)src[j];
    }
    {  // raw q tile (rows k0..k0+31 of qkv == head k0/32)
      const int kk = tid >> 3;
      const int pbq = (tid & 7) * 16;
      const _Float16* src =
          qkv + (((size_t)b * 768 + k0 + kk) << 12) + N0 + pbq;
#pragma unroll
      for (int j = 0; j < 16; ++j) Qt[kk * 128 + pbq + j] = (float)src[j];
    }
    __syncthreads();
    if (tid < 128) {  // per-pixel softmax over the 32 d-rows of this head
      const int p = tid;
      float m = -1e30f;
#pragma unroll
      for (int kk = 0; kk < 32; ++kk) m = fmaxf(m, Qt[kk * 128 + p]);
      float s = 0.f;
#pragma unroll
      for (int kk = 0; kk < 32; ++kk) s += __expf(Qt[kk * 128 + p] - m);
      const float r = qscale / s;
#pragma unroll
      for (int kk = 0; kk < 32; ++kk)
        Bs[p * 32 + kk] = (_Float16)(__expf(Qt[kk * 128 + p] - m) * r);
    }
    __syncthreads();

    v16h af[2], bf[4];
    af[0] = frag_a(As + (wm * 32 + 0) * 32, 32, lane);
    af[1] = frag_a(As + (wm * 32 + 16) * 32, 32, lane);
#pragma unroll
    for (int nt = 0; nt < 4; ++nt)
      bf[nt] = frag_b(Bs + (wn * 64 + nt * 16) * 32, 32, lane);
#pragma unroll
    for (int mt = 0; mt < 2; ++mt)
#pragma unroll
      for (int nt = 0; nt < 4; ++nt)
        acc[mt][nt] = wmma_f16(af[mt], bf[nt], acc[mt][nt]);
    __syncthreads();
  }

  const int mrow = (lane >> 4) * 8;
  const int ncol = lane & 15;
#pragma unroll
  for (int mt = 0; mt < 2; ++mt)
#pragma unroll
    for (int nt = 0; nt < 4; ++nt) {
      const int obase = M0 + wm * 32 + mt * 16 + mrow;
      const int p     = N0 + wn * 64 + nt * 16 + ncol;
      float* dst = y + (((size_t)b * 256 + obase) << 12) + p;
#pragma unroll
      for (int r = 0; r < 8; ++r)
        dst[(size_t)r << 12] = acc[mt][nt][r] + bout[obase + r];
    }
}

// ---------------------------------------------------------------------------
// K5: in-place channel LayerNorm * g. grid (16 b, 16 ptiles), one thread/pixel.
// ---------------------------------------------------------------------------
__global__ __launch_bounds__(256)
void layernorm(float* __restrict__ y, const float* __restrict__ g) {
  const int b = blockIdx.x;
  const int p = blockIdx.y * 256 + threadIdx.x;
  float* col = y + (((size_t)b * 256) << 12) + p;
  float s = 0.f, s2 = 0.f;
  for (int o = 0; o < 256; ++o) {
    const float v = col[(size_t)o << 12];
    s += v;
    s2 += v * v;
  }
  const float mean = s * (1.0f / 256.0f);
  const float var  = s2 * (1.0f / 256.0f) - mean * mean;
  const float rstd = rsqrtf(var + 1e-5f);
  for (int o = 0; o < 256; ++o) {
    const float v = col[(size_t)o << 12];
    col[(size_t)o << 12] = (v - mean) * rstd * g[o];
  }
}

// ---------------------------------------------------------------------------
extern "C" void kernel_launch(void* const* d_in, const int* in_sizes, int n_in,
                              void* d_out, int out_size, void* d_ws,
                              size_t ws_size, hipStream_t stream) {
  (void)in_sizes; (void)n_in; (void)out_size; (void)ws_size;
  const float* x    = (const float*)d_in[0];
  const float* wqkv = (const float*)d_in[1];
  const float* wout = (const float*)d_in[2];
  const float* bout = (const float*)d_in[3];
  const float* g    = (const float*)d_in[4];
  float* out = (float*)d_out;

  char* ws = (char*)d_ws;
  _Float16* qkv = (_Float16*)ws;                       // 100663296 B
  float* kmax   = (float*)(ws + 100663296);            // 16384 B
  float* ksum   = kmax + 4096;                         // 16384 B
  float* ctx    = ksum + 4096;                         // 524288 B
  float* w2     = ctx + 131072;                        // 4194304 B

  qkv_gemm<<<dim3(32, 6, 16), 256, 0, stream>>>(x, wqkv, qkv);
  kstat<<<4096, 256, 0, stream>>>(qkv, kmax, ksum);
  context_gemm<<<128, 256, 0, stream>>>(qkv, kmax, ksum, ctx);
  fold_w2<<<128, 256, 0, stream>>>(wout, ctx, w2);
  out_gemm<<<dim3(32, 2, 16), 256, 0, stream>>>(qkv, w2, bout, out);
  layernorm<<<dim3(16, 16), 256, 0, stream>>>(out, g);
}
